// LocalPointModel_77171972375068
// MI455X (gfx1250) — compile-verified, matrix operand order checked by reading, not run on hardware
//
#include <hip/hip_runtime.h>

typedef __attribute__((ext_vector_type(16))) __bf16 v16bf;
typedef __attribute__((ext_vector_type(8)))  __bf16 v8bf;
typedef __attribute__((ext_vector_type(8)))  float  v8f;

#define Bb 4
#define Nn 8192
#define BN (Bb * Nn)
#define KNN 16
#define Hh 128
#define Gg 256

// ---------------------------------------------------------------------------
// A-fragment load, 16-bit A matrix 16x32 layout (CDNA5 ISA 7.12.2):
// lane m=lane&15, hi=lane>>4: row m, K = {hi*8..hi*8+7, 16+hi*8..16+hi*8+7}
// => two contiguous 16-byte loads per lane (row-major A, stride in halves).
// ---------------------------------------------------------------------------
__device__ __forceinline__ v16bf afrag_load(const __bf16* chunk, int stride, int lane) {
  const int m = lane & 15;
  const int hiOff = (lane & 16) ? 8 : 0;
  const __bf16* p = chunk + (size_t)m * stride + hiOff;
  v8bf lo = *(const v8bf*)(p);
  v8bf hi = *(const v8bf*)(p + 16);
  v16bf a;
#pragma unroll
  for (int j = 0; j < 8; ++j) { a[j] = lo[j]; a[j + 8] = hi[j]; }
  return a;
}

// ---------------------------------------------------------------------------
// Convert f32 weight [K x N] row-major into B-fragment-major bf16 buffer:
// frag[(c*ntiles + nt)*512 + lane*16 + j], K padded to Kp with zeros.
// ---------------------------------------------------------------------------
__global__ void make_bfrag(const float* __restrict__ W, __bf16* __restrict__ frag,
                           int K, int Kp, int N) {
  long e = (long)blockIdx.x * blockDim.x + threadIdx.x;
  if (e >= (long)Kp * N) return;
  int j = (int)(e & 15);
  int lane = (int)((e >> 4) & 31);
  long fi = e >> 9;                      // fragment index = c*ntiles + nt
  int ntiles = N >> 4;
  int c  = (int)(fi / ntiles);
  int nt = (int)(fi % ntiles);
  const int hiOff = (lane & 16) ? 8 : 0;
  const int v = j >> 1, p = j & 1;
  const int kc = ((v >= 4) ? 16 : 0) + ((v & 3) << 1) + p + hiOff;
  int k = c * 32 + kc;
  int n = nt * 16 + (lane & 15);
  float val = (k < K) ? W[(size_t)k * N + n] : 0.f;
  frag[e] = (__bf16)val;
}

// ---------------------------------------------------------------------------
// x32 = [points(3) | normals(3) | zeros(26)] bf16, padded K for first GEMM
// ---------------------------------------------------------------------------
__global__ void build_x32(const float* __restrict__ pts, const float* __restrict__ nrm,
                          __bf16* __restrict__ x32) {
  long e = (long)blockIdx.x * blockDim.x + threadIdx.x;
  if (e >= (long)BN * 32) return;
  long row = e >> 5;
  int c = (int)(e & 31);
  float v = 0.f;
  if (c < 3)      v = pts[row * 3 + c];
  else if (c < 6) v = nrm[row * 3 + (c - 3)];
  x32[e] = (__bf16)v;
}

// ---------------------------------------------------------------------------
// Generic WMMA GEMM: C[M x N] = relu?(A[M x K](bf16,row-major,lda) * B + bias)
// One wave computes a 16 x (16*NT) tile. NT and NC (K-chunks) are compile-time
// so both loops fully unroll: loads pipeline across WMMAs, partial waits.
// ---------------------------------------------------------------------------
template <int NT, int NC>
__global__ __launch_bounds__(256) void gemm_wmma(
    const __bf16* __restrict__ A, int lda,
    const __bf16* __restrict__ Bfrag, const float* __restrict__ bias,
    float* __restrict__ outF, __bf16* __restrict__ outH,
    int ldc, int mtiles, int ntiles, int doRelu) {
  int wave = (int)(((long)blockIdx.x * blockDim.x + threadIdx.x) >> 5);
  int lane = threadIdx.x & 31;
  int ngroups = ntiles / NT;
  if (wave >= mtiles * ngroups) return;
  int mt  = wave % mtiles;
  int nt0 = (wave / mtiles) * NT;

  v8f acc[NT];
#pragma unroll
  for (int t = 0; t < NT; ++t) acc[t] = v8f{};
  const __bf16* Arow = A + (size_t)mt * 16 * lda;
  const __bf16* Bbase = Bfrag + ((size_t)nt0) * 512 + lane * 16;

#pragma unroll
  for (int c = 0; c < NC; ++c) {
    v16bf a = afrag_load(Arow + c * 32, lda, lane);
    const __bf16* bp = Bbase + (size_t)c * ntiles * 512;
#pragma unroll
    for (int t = 0; t < NT; ++t) {
      v16bf b = *(const v16bf*)(bp + (size_t)t * 512);
      acc[t] = __builtin_amdgcn_wmma_f32_16x16x32_bf16(
          false, a, false, b, (short)0, acc[t], false, false);
    }
  }

  // Epilogue: D layout — VGPR r: lanes0-15 M=r, lanes16-31 M=8+r; col=lane&15
  const int mlo = lane & 15;
  const int hi8 = (lane & 16) ? 8 : 0;
#pragma unroll
  for (int t = 0; t < NT; ++t) {
    int n = (nt0 + t) * 16 + mlo;
    float bv = bias ? bias[n] : 0.f;
#pragma unroll
    for (int r = 0; r < 8; ++r) {
      float vv = acc[t][r] + bv;
      if (doRelu) vv = fmaxf(vv, 0.f);
      size_t m = (size_t)mt * 16 + r + hi8;
      if (outF) outF[m * ldc + n] = vv;
      if (outH) outH[m * ldc + n] = (__bf16)vv;
    }
  }
}

// ---------------------------------------------------------------------------
// Brute-force kNN (k=16, excludes self), LDS-tiled, register insertion sort.
// Stores GLOBAL row indices (b*N + j).
// ---------------------------------------------------------------------------
__global__ __launch_bounds__(256) void knn_kernel(const float* __restrict__ pts,
                                                  int* __restrict__ knn) {
  int q = blockIdx.x * 256 + threadIdx.x;
  int b = q >> 13;
  int qi = q & (Nn - 1);
  const float* P = pts + (size_t)b * Nn * 3;
  float qx = P[qi * 3], qy = P[qi * 3 + 1], qz = P[qi * 3 + 2];

  __shared__ float tile[256 * 3];
  float bd[KNN]; int bi[KNN];
#pragma unroll
  for (int s = 0; s < KNN; ++s) { bd[s] = 3.0e38f; bi[s] = 0; }

  for (int t0 = 0; t0 < Nn; t0 += 256) {
    int j = t0 + threadIdx.x;
    tile[threadIdx.x * 3 + 0] = P[j * 3 + 0];
    tile[threadIdx.x * 3 + 1] = P[j * 3 + 1];
    tile[threadIdx.x * 3 + 2] = P[j * 3 + 2];
    __syncthreads();
    for (int jj = 0; jj < 256; ++jj) {
      int j2 = t0 + jj;
      float dx = tile[jj * 3 + 0] - qx;
      float dy = tile[jj * 3 + 1] - qy;
      float dz = tile[jj * 3 + 2] - qz;
      float d = dx * dx + dy * dy + dz * dz;
      if (j2 != qi && d < bd[KNN - 1]) {
#pragma unroll
        for (int s = KNN - 1; s >= 0; --s) {
          float pdv = (s > 0) ? bd[s - 1] : -1.0f;
          if (d < bd[s]) {
            if (s > 0 && d < pdv) { bd[s] = pdv; bi[s] = bi[s - 1]; }
            else                  { bd[s] = d;   bi[s] = j2; }
          }
        }
      }
    }
    __syncthreads();
  }
#pragma unroll
  for (int s = 0; s < KNN; ++s) knn[(size_t)q * KNN + s] = b * Nn + bi[s];
}

// ---------------------------------------------------------------------------
// Edge MLP: one wave per point. 16 neighbors -> M=16 tile.
// edge(16x262, pad 288) -> W_lm1(288x128) relu -> W_lm2(128x128) relu -> max_M
// Writes lf into feat columns [128,256).
// ---------------------------------------------------------------------------
#define WPB 4
__global__ __launch_bounds__(WPB * 32) void edge_kernel(
    const __bf16* __restrict__ feat,
    const float* __restrict__ pts, const float* __restrict__ nrm,
    const int* __restrict__ knn,
    const __bf16* __restrict__ w1f, const float* __restrict__ b1,
    const __bf16* __restrict__ w2f, const float* __restrict__ b2,
    __bf16* __restrict__ lfout /* = feat + 128 */) {
  __shared__ __align__(16) __bf16 eds[WPB][16 * 288];
  __shared__ __align__(16) __bf16 hds[WPB][16 * 128];
  int wv = threadIdx.x >> 5, lane = threadIdx.x & 31;
  size_t p = (size_t)blockIdx.x * WPB + wv;

  // -- build edge rows: 2 lanes per neighbor row m, each covering 131 cols
  int m = lane >> 1, half = lane & 1;
  int nb = knn[p * KNN + m];
  const __bf16* cfr = feat + p * 512;
  const __bf16* nfr = feat + (size_t)nb * 512;
  int k0 = half * 131;
  for (int k = k0; k < k0 + 131; ++k) {
    float val;
    if (k < 128)      val = (float)cfr[k];
    else if (k < 256) val = (float)nfr[k - 128] - (float)cfr[k - 128];
    else if (k < 259) val = pts[(size_t)nb * 3 + (k - 256)] - pts[p * 3 + (k - 256)];
    else              val = nrm[(size_t)nb * 3 + (k - 259)] - nrm[p * 3 + (k - 259)];
    eds[wv][m * 288 + k] = (__bf16)val;
  }
  if (half) {
    for (int k = 262; k < 288; ++k) eds[wv][m * 288 + k] = (__bf16)0.f;
  }
  __syncthreads();

  // -- layer 1: 9 K-chunks x 8 N-tiles (fully unrolled: loads pipeline)
  v8f acc[8];
#pragma unroll
  for (int t = 0; t < 8; ++t) acc[t] = v8f{};
#pragma unroll
  for (int c = 0; c < 9; ++c) {
    v16bf a = afrag_load(&eds[wv][c * 32], 288, lane);
#pragma unroll
    for (int nt = 0; nt < 8; ++nt) {
      v16bf bm = *(const v16bf*)(w1f + ((size_t)c * 8 + nt) * 512 + lane * 16);
      acc[nt] = __builtin_amdgcn_wmma_f32_16x16x32_bf16(
          false, a, false, bm, (short)0, acc[nt], false, false);
    }
  }
  // bias + relu -> hidden to LDS (row-major 16x128)
  const int mlo = lane & 15;
  const int hi8 = (lane & 16) ? 8 : 0;
#pragma unroll
  for (int nt = 0; nt < 8; ++nt) {
    float bv = b1[nt * 16 + mlo];
#pragma unroll
    for (int r = 0; r < 8; ++r) {
      float vv = fmaxf(acc[nt][r] + bv, 0.f);
      hds[wv][(r + hi8) * 128 + nt * 16 + mlo] = (__bf16)vv;
    }
  }
  __syncthreads();

  // -- layer 2: 4 K-chunks x 8 N-tiles
#pragma unroll
  for (int t = 0; t < 8; ++t) acc[t] = v8f{};
#pragma unroll
  for (int c = 0; c < 4; ++c) {
    v16bf a = afrag_load(&hds[wv][c * 32], 128, lane);
#pragma unroll
    for (int nt = 0; nt < 8; ++nt) {
      v16bf bm = *(const v16bf*)(w2f + ((size_t)c * 8 + nt) * 512 + lane * 16);
      acc[nt] = __builtin_amdgcn_wmma_f32_16x16x32_bf16(
          false, a, false, bm, (short)0, acc[nt], false, false);
    }
  }
  // bias + relu + max over M (8 VGPRs, then lane<->lane+16)
#pragma unroll
  for (int nt = 0; nt < 8; ++nt) {
    float bv = b2[nt * 16 + mlo];
    float mx = 0.f;                       // relu floor folded into max
#pragma unroll
    for (int r = 0; r < 8; ++r) mx = fmaxf(mx, acc[nt][r] + bv);
    mx = fmaxf(mx, __shfl_xor(mx, 16, 32));
    if (lane < 16) lfout[p * 512 + nt * 16 + lane] = (__bf16)mx;
  }
}

// ---------------------------------------------------------------------------
// Global max/mean over N for ctx (post-relu, >= 0)
// ---------------------------------------------------------------------------
__global__ void reduce_ctx(const float* __restrict__ ctx, float* __restrict__ gmax,
                           float* __restrict__ gmean) {
  int b = blockIdx.x, g = threadIdx.x;
  const float* base = ctx + (size_t)b * Nn * Gg + g;
  float mx = -3.0e38f, sm = 0.f;
  for (int n = 0; n < Nn; ++n) {
    float v = base[(size_t)n * Gg];
    mx = fmaxf(mx, v);
    sm += v;
  }
  gmax[b * Gg + g]  = mx;
  gmean[b * Gg + g] = sm * (1.0f / Nn);
}

__global__ void gp_kernel(const float* __restrict__ gmax, const float* __restrict__ gmean,
                          const float* __restrict__ Wgp, const float* __restrict__ bgp,
                          __bf16* __restrict__ gf) {
  int b = blockIdx.x, j = threadIdx.x;
  float s = bgp[j];
  for (int i = 0; i < Gg; ++i) s += gmax[b * Gg + i]  * Wgp[(size_t)i * Gg + j];
  for (int i = 0; i < Gg; ++i) s += gmean[b * Gg + i] * Wgp[(size_t)(Gg + i) * Gg + j];
  gf[b * Gg + j] = (__bf16)fmaxf(s, 0.f);
}

__global__ void bcast_gf(const __bf16* __restrict__ gf, __bf16* __restrict__ feat) {
  long e = (long)blockIdx.x * blockDim.x + threadIdx.x;   // BN * 256
  if (e >= (long)BN * Gg) return;
  long row = e >> 8;
  int g = (int)(e & 255);
  int b = (int)(row >> 13);
  feat[row * 512 + 256 + g] = gf[b * Gg + g];
}

__global__ void boundary_kernel(const __bf16* __restrict__ hid, const float* __restrict__ W,
                                const float* __restrict__ bb, float* __restrict__ out) {
  long row = (long)blockIdx.x * blockDim.x + threadIdx.x;
  if (row >= BN) return;
  float s = bb[0];
  for (int i = 0; i < Hh; ++i) s += (float)hid[row * Hh + i] * W[i];
  out[row] = s;
}

// ---------------------------------------------------------------------------
extern "C" void kernel_launch(void* const* d_in, const int* in_sizes, int n_in,
                              void* d_out, int out_size, void* d_ws, size_t ws_size,
                              hipStream_t stream) {
  const float* pts   = (const float*)d_in[0];
  const float* nrm   = (const float*)d_in[1];
  const float* W_pe1 = (const float*)d_in[2];  const float* b_pe1 = (const float*)d_in[3];
  const float* W_pe2 = (const float*)d_in[4];  const float* b_pe2 = (const float*)d_in[5];
  const float* W_lm1 = (const float*)d_in[6];  const float* b_lm1 = (const float*)d_in[7];
  const float* W_lm2 = (const float*)d_in[8];  const float* b_lm2 = (const float*)d_in[9];
  const float* W_cp  = (const float*)d_in[10]; const float* b_cp  = (const float*)d_in[11];
  const float* W_gp  = (const float*)d_in[12]; const float* b_gp  = (const float*)d_in[13];
  const float* W_cl1 = (const float*)d_in[14]; const float* b_cl1 = (const float*)d_in[15];
  const float* W_cl2 = (const float*)d_in[16]; const float* b_cl2 = (const float*)d_in[17];
  const float* W_ph1 = (const float*)d_in[18]; const float* b_ph1 = (const float*)d_in[19];
  const float* W_ph2 = (const float*)d_in[20]; const float* b_ph2 = (const float*)d_in[21];
  const float* W_bh1 = (const float*)d_in[22]; const float* b_bh1 = (const float*)d_in[23];
  const float* W_bh2 = (const float*)d_in[24]; const float* b_bh2 = (const float*)d_in[25];

  float* out_logits = (float*)d_out;                     // BN*32
  float* out_param  = out_logits + (size_t)BN * 32;      // BN*16
  float* out_bound  = out_param  + (size_t)BN * 16;      // BN

  // ---- workspace carve-out
  char* base = (char*)d_ws;
  size_t off = 0;
  auto alloc = [&](size_t bytes) { void* p = base + off; off = (off + bytes + 255) & ~(size_t)255; return p; };
  __bf16* feat  = (__bf16*)alloc((size_t)BN * 512 * 2);   // [pf | lf | gf]
  float*  ctx   = (float*) alloc((size_t)BN * Gg * 4);
  __bf16* hid   = (__bf16*)alloc((size_t)BN * Hh * 2);    // reused by heads
  __bf16* x32   = (__bf16*)alloc((size_t)BN * 32 * 2);
  int*    knn   = (int*)   alloc((size_t)BN * KNN * 4);
  float*  gmaxb = (float*) alloc(Bb * Gg * 4);
  float*  gmeanb= (float*) alloc(Bb * Gg * 4);
  __bf16* gf    = (__bf16*)alloc(Bb * Gg * 2);
  __bf16* fpe1  = (__bf16*)alloc((size_t)32  * 128 * 2);
  __bf16* fpe2  = (__bf16*)alloc((size_t)128 * 128 * 2);
  __bf16* flm1  = (__bf16*)alloc((size_t)288 * 128 * 2);
  __bf16* flm2  = (__bf16*)alloc((size_t)128 * 128 * 2);
  __bf16* fcp   = (__bf16*)alloc((size_t)256 * 256 * 2);
  __bf16* fcl1  = (__bf16*)alloc((size_t)512 * 128 * 2);
  __bf16* fcl2  = (__bf16*)alloc((size_t)128 * 32  * 2);
  __bf16* fph1  = (__bf16*)alloc((size_t)512 * 128 * 2);
  __bf16* fph2  = (__bf16*)alloc((size_t)128 * 16  * 2);
  __bf16* fbh1  = (__bf16*)alloc((size_t)512 * 128 * 2);
  (void)ws_size; (void)in_sizes; (void)n_in; (void)out_size;

  auto mkfrag = [&](const float* W, __bf16* f, int K, int Kp, int N) {
    long tot = (long)Kp * N;
    make_bfrag<<<(int)((tot + 255) / 256), 256, 0, stream>>>(W, f, K, Kp, N);
  };
  mkfrag(W_pe1, fpe1, 6,   32,  128);
  mkfrag(W_pe2, fpe2, 128, 128, 128);
  mkfrag(W_lm1, flm1, 262, 288, 128);
  mkfrag(W_lm2, flm2, 128, 128, 128);
  mkfrag(W_cp,  fcp,  256, 256, 256);
  mkfrag(W_cl1, fcl1, 512, 512, 128);
  mkfrag(W_cl2, fcl2, 128, 128, 32);
  mkfrag(W_ph1, fph1, 512, 512, 128);
  mkfrag(W_ph2, fph2, 128, 128, 16);
  mkfrag(W_bh1, fbh1, 512, 512, 128);

  // GEMM dispatch over compile-time (NT = n-tiles/wave, NC = K/32 chunks)
  auto gemm = [&](const __bf16* A, int lda, const __bf16* Bf, const float* bias,
                  float* oF, __bf16* oH, int ldc, int M, int K, int N, int relu) {
    int mtiles = M / 16, ntiles = N / 16, nch = K / 32;
    int NT = (ntiles % 4 == 0) ? 4 : ((ntiles % 2 == 0) ? 2 : 1);
    int ngroups = ntiles / NT;
    long waves = (long)mtiles * ngroups;
    int blocks = (int)((waves * 32 + 255) / 256);
#define LGEMM(NTV, NCV)                                                        \
    gemm_wmma<NTV, NCV><<<blocks, 256, 0, stream>>>(A, lda, Bf, bias, oF, oH,  \
                                                    ldc, mtiles, ntiles, relu)
    if      (NT == 4 && nch == 1)  LGEMM(4, 1);
    else if (NT == 4 && nch == 4)  LGEMM(4, 4);
    else if (NT == 4 && nch == 8)  LGEMM(4, 8);
    else if (NT == 4 && nch == 16) LGEMM(4, 16);
    else if (NT == 2 && nch == 4)  LGEMM(2, 4);
    else if (NT == 1 && nch == 4)  LGEMM(1, 4);
#undef LGEMM
  };

  // point encoder
  build_x32<<<(BN * 32 + 255) / 256, 256, 0, stream>>>(pts, nrm, x32);
  gemm(x32, 32,  fpe1, b_pe1, nullptr, hid,  Hh,  BN, 32,  Hh, 1);
  gemm(hid, Hh,  fpe2, b_pe2, nullptr, feat, 512, BN, Hh,  Hh, 1);   // pf -> feat[:,0:128]

  // kNN + edge MLP (lf -> feat[:,128:256])
  knn_kernel<<<BN / 256, 256, 0, stream>>>(pts, knn);
  edge_kernel<<<BN / WPB, WPB * 32, 0, stream>>>(feat, pts, nrm, knn,
                                                 flm1, b_lm1, flm2, b_lm2, feat + 128);

  // ctx -> global pooling -> gf broadcast (feat[:,256:512])
  gemm(feat, 512, fcp, b_cp, ctx, nullptr, Gg, BN, 256, Gg, 1);
  reduce_ctx<<<Bb, Gg, 0, stream>>>(ctx, gmaxb, gmeanb);
  gp_kernel<<<Bb, Gg, 0, stream>>>(gmaxb, gmeanb, W_gp, b_gp, gf);
  bcast_gf<<<(BN * Gg + 255) / 256, 256, 0, stream>>>(gf, feat);

  // heads (hid buffer reused sequentially; stream order serializes)
  gemm(feat, 512, fcl1, b_cl1, nullptr, hid, Hh, BN, 512, Hh, 1);
  gemm(hid,  Hh,  fcl2, b_cl2, out_logits, nullptr, 32, BN, Hh, 32, 0);
  gemm(feat, 512, fph1, b_ph1, nullptr, hid, Hh, BN, 512, Hh, 1);
  gemm(hid,  Hh,  fph2, b_ph2, out_param, nullptr, 16, BN, Hh, 16, 0);
  gemm(feat, 512, fbh1, b_bh1, nullptr, hid, Hh, BN, 512, Hh, 1);
  boundary_kernel<<<(BN + 255) / 256, 256, 0, stream>>>(hid, W_bh2, b_bh2, out_bound);
}
